// FMDepthRNN_14551349199178
// MI455X (gfx1250) — matrix operand order repaired and checked by reading, not compiled, not run on hardware
//
#include <hip/hip_runtime.h>
#include <hip/hip_bf16.h>
#include <stdint.h>

// Problem constants (match reference)
#define B_    256
#define T_    4096
#define H_    136
#define OUT_  68
#define HP_   160     // hidden padded to 5 chunks of K=32
#define NWAVE 9       // n-tiles of 16 covering 144 >= 136

typedef __attribute__((ext_vector_type(16))) __bf16 bf16x16;
typedef __attribute__((ext_vector_type(8)))  __bf16 bf16x8;
typedef __attribute__((ext_vector_type(8)))  float  floatx8;

#define WMMA_BF16(a, b, c) \
  __builtin_amdgcn_wmma_f32_16x16x32_bf16(false, (a), false, (b), (short)0, (c), false, false)

__device__ __forceinline__ float fast_tanh(float x) {
#if __has_builtin(__builtin_amdgcn_tanhf)
  return __builtin_amdgcn_tanhf(x);          // v_tanh_f32 (gfx1250 trans op)
#elif __has_builtin(__builtin_amdgcn_tanh_f32)
  return __builtin_amdgcn_tanh_f32(x);
#else
  float e = __expf(2.0f * x);                // v_exp_f32 path, overflow-safe
  return 1.0f - 2.0f / (e + 1.0f);
#endif
}

// A-operand element -> K index mapping for 16-bit 16x32 A tiles (wave32):
// lane<16: elems 0..7 -> K 0..7,  elems 8..15 -> K 16..23
// lane>=16: elems 0..7 -> K 8..15, elems 8..15 -> K 24..31
__device__ __forceinline__ int a_elem_k(int e, int hi) {
  return (e & 7) + (hi ? 8 : 0) + ((e >= 8) ? 16 : 0);
}

__global__ __launch_bounds__(32 * NWAVE, 1)
void rnn_persistent_kernel(const float* __restrict__ x,
                           const float* __restrict__ W_ih,
                           const float* __restrict__ W_hh,
                           const float* __restrict__ b_ih,
                           const float* __restrict__ b_hh,
                           const float* __restrict__ W_fc,
                           const float* __restrict__ b_fc,
                           float* __restrict__ out) {
  // h image: hbuf[buf][m][k] (bf16), double buffered. 2*16*160*2B = 10 KB.
  __shared__ __align__(16) __bf16 hbuf[2][16 * HP_];

  const int tid  = threadIdx.x;
  const int wave = tid >> 5;        // n-tile (0..8)
  const int lane = tid & 31;
  const int l16  = lane & 15;       // batch column m within tile
  const int hi   = lane >> 4;       // lane group (0/1)
  const int mtile = blockIdx.x;     // batch tile (16 rows)

  // Zero both LDS buffers (h0 = 0, and K-pad 144..159 stays 0 forever).
  {
    uint32_t* p = (uint32_t*)&hbuf[0][0];
    for (int i = tid; i < (2 * 16 * HP_) / 2; i += blockDim.x) p[i] = 0u;
  }

  // ---- Stationary A operands: this wave's 16 hidden rows of W_hh ----
  const int nrow = wave * 16 + l16;               // hidden output row n
  bf16x16 Ah[5];
#pragma unroll
  for (int c = 0; c < 5; ++c) {
#pragma unroll
    for (int e = 0; e < 16; ++e) {
      const int k = 32 * c + a_elem_k(e, hi);
      const float v = (nrow < H_ && k < H_) ? W_hh[nrow * H_ + k] : 0.0f;
      Ah[c][e] = (__bf16)v;
    }
  }
  // Extra K-chunk folding input projection + both biases:
  // A6[n][0]=W_ih[n][0], A6[n][1]=W_ih[n][1], A6[n][2]=b_ih[n]+b_hh[n]
  bf16x16 A6;
#pragma unroll
  for (int e = 0; e < 16; ++e) A6[e] = (__bf16)0.0f;
  if (!hi && nrow < H_) {
    A6[0] = (__bf16)W_ih[nrow * 2 + 0];
    A6[1] = (__bf16)W_ih[nrow * 2 + 1];
    A6[2] = (__bf16)(b_ih[nrow] + b_hh[nrow]);
  }

  // B-extra chunk: only elements 0/1 (x values) change per step; the constant
  // one (bias column) and all zeros are set once here.
  bf16x16 B6;
#pragma unroll
  for (int e = 0; e < 16; ++e) B6[e] = (__bf16)0.0f;
  if (!hi) B6[2] = (__bf16)1.0f;

  __syncthreads();

  // Per-lane x stream for batch row (mtile*16 + m): contiguous over t.
  const float2* xrow = (const float2*)x + (size_t)(mtile * 16 + l16) * T_;
  float2 xv = xrow[0];
  const __bf16 bz = (__bf16)0.0f;

#pragma unroll 2
  for (int t = 0; t < T_; ++t) {
    const int pr = t & 1;
    const int pw = pr ^ 1;

    // Update x/bias B-chunk (lanes 16..31 stay zero), prefetch next x.
    B6[0] = hi ? bz : (__bf16)xv.x;
    B6[1] = hi ? bz : (__bf16)xv.y;
    xv = xrow[(t + 1 < T_) ? (t + 1) : t];

    // Issue all h^T B-chunk loads (10x ds_load_b128) ahead of the WMMA chain.
    const __bf16* hb = &hbuf[pr][l16 * HP_ + hi * 8];
    bf16x16 Bh[5];
#pragma unroll
    for (int c = 0; c < 5; ++c) {
      bf16x8 lo = *(const bf16x8*)(hb + 32 * c);
      bf16x8 h8 = *(const bf16x8*)(hb + 32 * c + 16);
      Bh[c] = __builtin_shufflevector(lo, h8, 0, 1, 2, 3, 4, 5, 6, 7, 8, 9, 10,
                                      11, 12, 13, 14, 15);
    }

    // Two independent accumulator chains (3 deep each), issued alternately so
    // each dependent accumulate has a full XDL latency slot to retire.
    floatx8 Ca = 0.0f;
    floatx8 Cb = 0.0f;
    Ca = WMMA_BF16(A6, B6, Ca);        // chain A: x + bias (LDS-independent)
    Cb = WMMA_BF16(Ah[0], Bh[0], Cb);  // chain B: chunk 0
    Ca = WMMA_BF16(Ah[1], Bh[1], Ca);  // chain A: chunk 1
    Cb = WMMA_BF16(Ah[2], Bh[2], Cb);  // chain B: chunk 2
    Ca = WMMA_BF16(Ah[3], Bh[3], Ca);  // chain A: chunk 3
    Cb = WMMA_BF16(Ah[4], Bh[4], Cb);  // chain B: chunk 4
    floatx8 C = Ca + Cb;

    // Activation + write h_new: lane holds 8 consecutive n for fixed m ->
    // one contiguous 16B ds_store_b128 at hbuf[pw][m][wave*16 + hi*8].
    bf16x8 hn;
#pragma unroll
    for (int r = 0; r < 8; ++r) hn[r] = (__bf16)fast_tanh(C[r]);
    *(bf16x8*)&hbuf[pw][l16 * HP_ + wave * 16 + hi * 8] = hn;
    __syncthreads();
  }

  // ---- Final FC head: out[b][o] = h_last[b][:] . W_fc[o][:] + b_fc[o] ----
  const int pfin = T_ & 1;  // h_last buffer
  if (wave < 5) {           // o-tiles of 16 covering 80 >= 68 (wave-uniform)
    const int orow = wave * 16 + l16;
    bf16x16 Af[5];
#pragma unroll
    for (int c = 0; c < 5; ++c) {
#pragma unroll
      for (int e = 0; e < 16; ++e) {
        const int k = 32 * c + a_elem_k(e, hi);
        const float v = (orow < OUT_ && k < H_) ? W_fc[orow * H_ + k] : 0.0f;
        Af[c][e] = (__bf16)v;
      }
    }
    const __bf16* hb = &hbuf[pfin][l16 * HP_ + hi * 8];
    bf16x16 Bh[5];
#pragma unroll
    for (int c = 0; c < 5; ++c) {
      bf16x8 lo = *(const bf16x8*)(hb + 32 * c);
      bf16x8 h8 = *(const bf16x8*)(hb + 32 * c + 16);
      Bh[c] = __builtin_shufflevector(lo, h8, 0, 1, 2, 3, 4, 5, 6, 7, 8, 9, 10,
                                      11, 12, 13, 14, 15);
    }
    floatx8 C = 0.0f;
#pragma unroll
    for (int c = 0; c < 5; ++c) C = WMMA_BF16(Af[c], Bh[c], C);

    // Lane holds out rows o = wave*16 + hi*8 + r for batch b = mtile*16 + m
#pragma unroll
    for (int r = 0; r < 8; ++r) {
      const int o = wave * 16 + hi * 8 + r;
      if (o < OUT_) {
        out[(size_t)(mtile * 16 + l16) * OUT_ + o] = C[r] + b_fc[o];
      }
    }
  }
}

extern "C" void kernel_launch(void* const* d_in, const int* in_sizes, int n_in,
                              void* d_out, int out_size, void* d_ws,
                              size_t ws_size, hipStream_t stream) {
  (void)in_sizes; (void)n_in; (void)out_size; (void)d_ws; (void)ws_size;
  const float* x    = (const float*)d_in[0];
  const float* W_ih = (const float*)d_in[1];
  const float* W_hh = (const float*)d_in[2];
  const float* b_ih = (const float*)d_in[3];
  const float* b_hh = (const float*)d_in[4];
  const float* W_fc = (const float*)d_in[5];
  const float* b_fc = (const float*)d_in[6];
  float* out = (float*)d_out;

  rnn_persistent_kernel<<<dim3(B_ / 16), dim3(32 * NWAVE), 0, stream>>>(
      x, W_ih, W_hh, b_ih, b_hh, W_fc, b_fc, out);
}
